// Sequence_1022202217225
// MI455X (gfx1250) — matrix operand order repaired and checked by reading, not compiled, not run on hardware
//
#include <hip/hip_runtime.h>
#include <hip/hip_bf16.h>
#include <math.h>

typedef __attribute__((ext_vector_type(16))) _Float16 v16h;
typedef __attribute__((ext_vector_type(8)))  float    v8f;

// ---------------- problem constants ----------------
#define T_OBS 4096
#define H1 151
#define H2 251
#define H3 51

// padded geometry (gates padded to 16, K padded to 32)
#define G1P 608           // 4*151=604 -> 608  (38 gate tiles)
#define G2P 1008          // 4*251=1004 -> 1008 (63 gate tiles)
#define G3P 208           // 4*51=204 -> 208   (13 gate tiles)
#define K1P 160           // 2+151=153 -> 160  (5 k tiles)   vec1 = [x(2); h1]
#define K2P 416           // 151+251=402 -> 416 (13 k tiles) vec2 = [h1; h2]
#define K3P 320           // 251+51=302 -> 320 (10 k tiles)  vec3 = [h2; h3]

#define NT1 38
#define KT1 5
#define NT2 63
#define KT2 13
#define NT3 13
#define KT3 10

#define TILES1 (NT1*KT1)                  // 190
#define TILES2 (NT2*KT2)                  // 819
#define TILES3 (NT3*KT3)                  // 130
#define TILES_TOTAL (TILES1+TILES2+TILES3) // 1139
#define A_HALVES (TILES_TOTAL*512)        // 583168 f16 elements
#define L1_OFF 0
#define L2_OFF (TILES1*512)               // 97280
#define L3_OFF ((TILES1+TILES2)*512)      // 516608
#define BIAS_TOTAL (G1P+G2P+G3P)          // 1824

// ---------------- prep: swizzle weights into WMMA A-operand layout + fuse biases ----------------
__device__ __forceinline__ float wcat_elem(
    int layer, int m, int k,
    const float* __restrict__ Wih1, const float* __restrict__ Whh1,
    const float* __restrict__ Wih2, const float* __restrict__ Whh2,
    const float* __restrict__ Wih3, const float* __restrict__ Whh3)
{
  if (layer == 0) {
    if (m >= 604) return 0.0f;
    if (k < 2)    return Wih1[m * 2 + k];
    if (k < 153)  return Whh1[m * 151 + (k - 2)];
    return 0.0f;
  } else if (layer == 1) {
    if (m >= 1004) return 0.0f;
    if (k < 151)   return Wih2[m * 151 + k];
    if (k < 402)   return Whh2[m * 251 + (k - 151)];
    return 0.0f;
  } else {
    if (m >= 204) return 0.0f;
    if (k < 251)  return Wih3[m * 251 + k];
    if (k < 302)  return Whh3[m * 51 + (k - 251)];
    return 0.0f;
  }
}

__global__ void lstm_prep_kernel(
    const float* __restrict__ Wih1, const float* __restrict__ Whh1,
    const float* __restrict__ bih1, const float* __restrict__ bhh1,
    const float* __restrict__ Wih2, const float* __restrict__ Whh2,
    const float* __restrict__ bih2, const float* __restrict__ bhh2,
    const float* __restrict__ Wih3, const float* __restrict__ Whh3,
    const float* __restrict__ bih3, const float* __restrict__ bhh3,
    _Float16* __restrict__ Aout, float* __restrict__ biasOut)
{
  int tid = blockIdx.x * blockDim.x + threadIdx.x;
  if (tid < A_HALVES) {
    // element index = tile*512 + lane*16 + j  (one v16h A fragment per lane)
    int tile = tid >> 9;
    int w    = tid & 511;
    int lane = w >> 4;
    int j    = w & 15;
    int layer, gt, kt;
    if (tile < TILES1)                { layer = 0; gt = tile / KT1;              kt = tile % KT1; }
    else if (tile < TILES1 + TILES2)  { int u = tile - TILES1; layer = 1; gt = u / KT2; kt = u % KT2; }
    else                              { int u = tile - TILES1 - TILES2; layer = 2; gt = u / KT3; kt = u % KT3; }
    // 16-bit A 16x32 layout (contiguous k order, matching B): lane<16 holds K 0..15 (j0-15);
    // lane>=16 holds K 16..31.
    int k = kt * 32 + j + ((lane >= 16) ? 16 : 0);
    int m = gt * 16 + (lane & 15);
    Aout[tid] = (_Float16)wcat_elem(layer, m, k, Wih1, Whh1, Wih2, Whh2, Wih3, Whh3);
  } else if (tid < A_HALVES + BIAS_TOTAL) {
    int g = tid - A_HALVES;
    float v;
    if (g < G1P)            v = (g < 604) ? (bih1[g] + bhh1[g]) : 0.0f;
    else if (g < G1P + G2P) { int q = g - G1P;       v = (q < 1004) ? (bih2[q] + bhh2[q]) : 0.0f; }
    else                    { int q = g - G1P - G2P; v = (q < 204)  ? (bih3[q] + bhh3[q]) : 0.0f; }
    biasOut[g] = v;
  }
}

// ---------------- persistent LSTM kernel ----------------
__device__ __forceinline__ float sigf(float v) { return 1.0f / (1.0f + expf(-v)); }

// gates[0..ntiles*16) = bias + W_swizzled * vec   (per-wave gate tiles, K-accumulated in C)
__device__ __forceinline__ void layer_wmma(
    const _Float16* __restrict__ Aw,     // swizzled weights, tile-major
    const float*    __restrict__ biasL,  // padded fused bias (LDS)
    const _Float16* __restrict__ vec,    // padded activation vector (LDS, f16)
    float*          __restrict__ gates,  // output gates (LDS, f32)
    int ntiles, int kt_cnt, int wave, int lane)
{
  const int halfsel = lane >> 4;                 // 0: M rows 0-7 / K lo, 1: M rows 8-15 / K hi
  for (int gt = wave; gt < ntiles; gt += 32) {
    const int gb = gt * 16 + halfsel * 8;
    v8f c;
#pragma unroll
    for (int r = 0; r < 8; ++r) c[r] = biasL[gb + r];
    const _Float16* ap = Aw + (size_t)gt * kt_cnt * 512 + lane * 16;
    const _Float16* vp = vec + halfsel * 16;
    for (int kt = 0; kt < kt_cnt; ++kt) {
      v16h a = *(const v16h*)ap;                 // 32B/lane, pre-swizzled global (L2-resident)
      v16h b = *(const v16h*)(vp + kt * 32);     // broadcast vector fragment from LDS
      c = __builtin_amdgcn_wmma_f32_16x16x32_f16(false, a, false, b, (short)0, c, false, false);
      ap += 512;
    }
    if ((lane & 15) == 0) {                      // lane 0 -> M 0..7, lane 16 -> M 8..15
#pragma unroll
      for (int r = 0; r < 8; ++r) gates[gb + r] = c[r];
    }
  }
}

__global__ __launch_bounds__(1024) void lstm_persist_kernel(
    const float* __restrict__ x,        // (T,2)
    const _Float16* __restrict__ Aw,    // swizzled weights (ws)
    const float* __restrict__ biasg,    // fused padded biases (ws)
    const float* __restrict__ Wl,       // (2,51)
    const float* __restrict__ bl,       // (2,)
    float* __restrict__ out,            // (T+future, 2)
    int total_steps)
{
  __shared__ __align__(32) _Float16 vec1[K1P];
  __shared__ __align__(32) _Float16 vec2[K2P];
  __shared__ __align__(32) _Float16 vec3[K3P];
  __shared__ float gates[G2P];          // reused by all three layers (max 1008)
  __shared__ float h1s[H1], c1s[H1], h2s[H2], c2s[H2], h3s[H3], c3s[H3];
  __shared__ float biasl[BIAS_TOTAL];
  __shared__ float wll[2 * 51];
  __shared__ float bll[2];
  __shared__ float ycur[2];

  const int tid  = threadIdx.x;
  const int lane = tid & 31;
  const int wave = tid >> 5;

  // ---- init ----
  for (int i = tid; i < BIAS_TOTAL; i += 1024) biasl[i] = biasg[i];
  if (tid < H1) { h1s[tid] = 0.0f; c1s[tid] = 0.0f; }
  if (tid < H2) { h2s[tid] = 0.0f; c2s[tid] = 0.0f; }
  if (tid < H3) { h3s[tid] = 0.0f; c3s[tid] = 0.0f; }
  if (tid < K1P) vec1[tid] = (_Float16)0.0f;
  if (tid < K2P) vec2[tid] = (_Float16)0.0f;
  if (tid < K3P) vec3[tid] = (_Float16)0.0f;
  if (tid < 102) wll[tid] = Wl[tid];
  if (tid < 2)  { bll[tid] = bl[tid]; ycur[tid] = 0.0f; }
  __syncthreads();

  for (int t = 0; t < total_steps; ++t) {
    // P0: feed input (observed x, or fed-back y in the future phase)
    if (tid < 2) vec1[tid] = (_Float16)((t < T_OBS) ? x[t * 2 + tid] : ycur[tid]);
    __syncthreads();

    // P1: layer-1 gates = [Wih1|Whh1] * [x; h1] + b
    layer_wmma(Aw + L1_OFF, biasl, vec1, gates, NT1, KT1, wave, lane);
    __syncthreads();

    // P2: update h1/c1, stage vec2 = [h1_new ; h2_old], stage vec1 h-part for next step
    if (tid < H1) {
      float ig = sigf(gates[tid]);
      float fg = sigf(gates[H1 + tid]);
      float gg = tanhf(gates[2 * H1 + tid]);
      float og = sigf(gates[3 * H1 + tid]);
      float cn = fg * c1s[tid] + ig * gg;
      float hn = og * tanhf(cn);
      c1s[tid] = cn; h1s[tid] = hn;
      _Float16 hh = (_Float16)hn;
      vec2[tid] = hh;
      vec1[2 + tid] = hh;
    } else if (tid < H1 + H2) {
      vec2[tid] = (_Float16)h2s[tid - H1];
    }
    __syncthreads();

    // P3: layer-2 gates
    layer_wmma(Aw + L2_OFF, biasl + G1P, vec2, gates, NT2, KT2, wave, lane);
    __syncthreads();

    // P4: update h2/c2, stage vec3 = [h2_new ; h3_old]
    if (tid < H2) {
      float ig = sigf(gates[tid]);
      float fg = sigf(gates[H2 + tid]);
      float gg = tanhf(gates[2 * H2 + tid]);
      float og = sigf(gates[3 * H2 + tid]);
      float cn = fg * c2s[tid] + ig * gg;
      float hn = og * tanhf(cn);
      c2s[tid] = cn; h2s[tid] = hn;
      vec3[tid] = (_Float16)hn;
    } else if (tid < H2 + H3) {
      vec3[tid] = (_Float16)h3s[tid - H2];
    }
    __syncthreads();

    // P5: layer-3 gates
    layer_wmma(Aw + L3_OFF, biasl + G1P + G2P, vec3, gates, NT3, KT3, wave, lane);
    __syncthreads();

    // P6: update h3/c3
    if (tid < H3) {
      float ig = sigf(gates[tid]);
      float fg = sigf(gates[H3 + tid]);
      float gg = tanhf(gates[2 * H3 + tid]);
      float og = sigf(gates[3 * H3 + tid]);
      float cn = fg * c3s[tid] + ig * gg;
      float hn = og * tanhf(cn);
      c3s[tid] = cn; h3s[tid] = hn;
    }
    __syncthreads();

    // P7: y = Wl * h3 + bl ; emit and stage for feedback
    if (tid < 2) {
      float acc = bll[tid];
#pragma unroll 3
      for (int j = 0; j < 51; ++j) acc += wll[tid * 51 + j] * h3s[j];
      out[t * 2 + tid] = acc;
      ycur[tid] = acc;
    }
    __syncthreads();
  }
}

// ---------------- host launcher ----------------
extern "C" void kernel_launch(void* const* d_in, const int* in_sizes, int n_in,
                              void* d_out, int out_size, void* d_ws, size_t ws_size,
                              hipStream_t stream) {
  (void)n_in; (void)in_sizes; (void)ws_size;
  const float* x    = (const float*)d_in[0];
  const float* Wih1 = (const float*)d_in[1];
  const float* Whh1 = (const float*)d_in[2];
  const float* bih1 = (const float*)d_in[3];
  const float* bhh1 = (const float*)d_in[4];
  const float* Wih2 = (const float*)d_in[5];
  const float* Whh2 = (const float*)d_in[6];
  const float* bih2 = (const float*)d_in[7];
  const float* bhh2 = (const float*)d_in[8];
  const float* Wih3 = (const float*)d_in[9];
  const float* Whh3 = (const float*)d_in[10];
  const float* bih3 = (const float*)d_in[11];
  const float* bhh3 = (const float*)d_in[12];
  const float* Wl   = (const float*)d_in[13];
  const float* bl   = (const float*)d_in[14];

  _Float16* Aw     = (_Float16*)d_ws;
  float*    biasws = (float*)((char*)d_ws + (size_t)A_HALVES * sizeof(_Float16));

  const int total_steps = out_size / 2;   // T + future (= 5120)

  int prep_threads = A_HALVES + BIAS_TOTAL;
  int prep_blocks  = (prep_threads + 255) / 256;
  lstm_prep_kernel<<<prep_blocks, 256, 0, stream>>>(
      Wih1, Whh1, bih1, bhh1, Wih2, Whh2, bih2, bhh2,
      Wih3, Whh3, bih3, bhh3, Aw, biasws);

  lstm_persist_kernel<<<1, 1024, 0, stream>>>(
      x, Aw, biasws, Wl, bl, (float*)d_out, total_steps);
}
